// QuotRemLinear_15324443312231
// MI455X (gfx1250) — matrix operand order repaired and checked by reading, not compiled, not run on hardware
//
#include <hip/hip_runtime.h>

// QuotRem fake-quant + linear, fused.  GEMM M=65536 N=1024 K=1024, f16 WMMA / f32 acc.
// Pass 1: w[n][k] f32 -> Wt[k][n] f16 (2 MB, L2-resident).
// Pass 2: fused quant(x)->LDS + WMMA GEMM; B tile staged via async global->LDS copies.

typedef __attribute__((ext_vector_type(16))) _Float16 v16h;
typedef __attribute__((ext_vector_type(8)))  _Float16 v8h;
typedef __attribute__((ext_vector_type(8)))  float    v8f;
typedef __attribute__((ext_vector_type(4)))  int      v4i;

#define KDIM     1024
#define NDIM     1024
#define GS       128      // quant group == K-chunk
#define MT       64
#define NT       512
#define THREADS  512      // 16 waves
#define AS_STR   136      // f16 row stride of A tile (272B, 16B aligned)
#define BT_STR   520      // f16 row stride of Bt tile (1040B, 16B aligned)
#define EPSQ     1e-8f

#define AS1 __attribute__((address_space(1)))
#define AS3 __attribute__((address_space(3)))

// generic->global: value-preserving.  generic->LDS: low 32 bits are the LDS offset
// (ISA 10.2 aperture table).
static __device__ __forceinline__ AS1 v4i* to_glob4(const void* p) {
    return (AS1 v4i*)(unsigned long long)(uintptr_t)p;
}
static __device__ __forceinline__ AS3 v4i* to_lds4(void* p) {
    return (AS3 v4i*)(unsigned int)(uintptr_t)p;
}

static __device__ __forceinline__ void copy16_g2lds(const void* g, void* l) {
#if __has_builtin(__builtin_amdgcn_global_load_async_to_lds_b128)
    __builtin_amdgcn_global_load_async_to_lds_b128(to_glob4(g), to_lds4(l), 0, 0);
#elif __has_builtin(__builtin_amdgcn_load_to_lds)
    __builtin_amdgcn_load_to_lds(to_glob4(g), to_lds4(l), 16, 0, 0);
#else
    *(v8h*)l = *(const v8h*)g;
#endif
}

static __device__ __forceinline__ void wait_async_copies() {
#if __has_builtin(__builtin_amdgcn_global_load_async_to_lds_b128) || __has_builtin(__builtin_amdgcn_load_to_lds)
#if __has_builtin(__builtin_amdgcn_s_wait_asynccnt)
    __builtin_amdgcn_s_wait_asynccnt(0);
#endif
#endif
}

// ---------------- pass 1: weight convert + transpose ----------------
__global__ __launch_bounds__(256)
void wt_transpose(const float* __restrict__ w, _Float16* __restrict__ wt)
{
    __shared__ __align__(16) _Float16 T[64 * 72];
    const int n0 = blockIdx.x * 64;
    const int k0 = blockIdx.y * 64;
    const int tid = threadIdx.x;

    {   // coalesced read w[n][k], transposed f16 store into LDS
        const int rr = tid >> 2;          // n_local 0..63
        const int c4 = tid & 3;           // 16-wide k segment
        const float4* src = (const float4*)(w + (size_t)(n0 + rr) * KDIM + k0 + c4 * 16);
#pragma unroll
        for (int j = 0; j < 4; ++j) {
            float4 v = src[j];
            int kk = c4 * 16 + j * 4;
            T[(kk + 0) * 72 + rr] = (_Float16)v.x;
            T[(kk + 1) * 72 + rr] = (_Float16)v.y;
            T[(kk + 2) * 72 + rr] = (_Float16)v.z;
            T[(kk + 3) * 72 + rr] = (_Float16)v.w;
        }
    }
    __syncthreads();
    {   // coalesced write Wt[k][n]
        const int kk = tid >> 2;          // k_local 0..63
        const int q  = tid & 3;           // 16-wide n segment
        v8h* dst = (v8h*)(wt + (size_t)(k0 + kk) * NDIM + n0 + q * 16);
        dst[0] = *(const v8h*)(T + kk * 72 + q * 16);
        dst[1] = *(const v8h*)(T + kk * 72 + q * 16 + 8);
    }
}

// ---------------- pass 2: fused quant + WMMA GEMM ----------------
template<bool USE_WT>
__global__ __launch_bounds__(THREADS)
void qr_linear_wmma(const float* __restrict__ x,
                    const float* __restrict__ w,
                    const _Float16* __restrict__ wt,
                    const float* __restrict__ bias,
                    float* __restrict__ out)
{
    const int tid  = threadIdx.x;
    const int lane = tid & 31;
    const int wave = tid >> 5;            // 0..15

    const int n_blk = blockIdx.x * NT;    // n fastest-varying -> x-tile L2 reuse
    const int m_blk = blockIdx.y * MT;

    __shared__ __align__(16) _Float16 As[MT * AS_STR];   // 17.4 KB
    __shared__ __align__(16) _Float16 Bt[GS * BT_STR];   // 133 KB

    // A-staging role: 8 threads per row, 16 elements each (group of 128 = one quant group)
    const int row = tid >> 3;             // 0..63
    const int seg = tid & 7;              // 0..7
    const float* xrow = x + (size_t)(m_blk + row) * KDIM + seg * 16;

    // B-staging role
    const int bk = tid >> 2;              // 0..127 (k within chunk)
    const int bq = tid & 3;               // 128-wide n quarter

    // compute role: wave = (mw, nw); 2 M-subtiles x 4 N-subtiles per wave
    const int mw    = wave >> 3;          // 0..1
    const int nw    = wave & 7;           // 0..7
    const int mOff0 = mw * 32;
    const int mOff1 = mOff0 + 16;
    const int nBase = nw * 64;
    const int lhalf = lane >> 4;
    const int lmod  = lane & 15;

    v8f acc[2][4] = {};

#pragma unroll 1
    for (int kc = 0; kc < KDIM / GS; ++kc) {
        // ---- stage B first: async copies overlap with the quant VALU below ----
        if constexpr (USE_WT) {
            const _Float16* bsrc = wt + (size_t)(kc * GS + bk) * NDIM + n_blk + bq * 128;
            _Float16* bdst = Bt + bk * BT_STR + bq * 128;
#pragma unroll
            for (int j = 0; j < 16; ++j)
                copy16_g2lds(bsrc + j * 8, bdst + j * 8);
        } else {
            // fallback: convert+transpose on the fly
            const float4* wp = (const float4*)(w + (size_t)(n_blk + tid) * KDIM + kc * GS);
#pragma unroll
            for (int j = 0; j < 32; ++j) {
                float4 v = wp[j];
                int k0 = j * 4;
                Bt[(k0 + 0) * BT_STR + tid] = (_Float16)v.x;
                Bt[(k0 + 1) * BT_STR + tid] = (_Float16)v.y;
                Bt[(k0 + 2) * BT_STR + tid] = (_Float16)v.z;
                Bt[(k0 + 3) * BT_STR + tid] = (_Float16)v.w;
            }
        }

        // ---- stage A: quotient-remainder quantization (group == this 128-chunk) ----
        {
            float xv[16], rv[16], qb[16];
            const float4* xp = (const float4*)(xrow + kc * GS);
            float amax = 0.f;
#pragma unroll
            for (int j = 0; j < 4; ++j) {
                float4 v = xp[j];
                xv[4*j+0] = v.x; xv[4*j+1] = v.y; xv[4*j+2] = v.z; xv[4*j+3] = v.w;
                amax = fmaxf(amax, fmaxf(fmaxf(fabsf(v.x), fabsf(v.y)),
                                         fmaxf(fabsf(v.z), fabsf(v.w))));
            }
            amax = fmaxf(amax, __shfl_xor(amax, 1, 32));
            amax = fmaxf(amax, __shfl_xor(amax, 2, 32));
            amax = fmaxf(amax, __shfl_xor(amax, 4, 32));
            amax = fmaxf(amax, EPSQ);

            const float thr = amax * (1.0f / 7.0f);
            float base = 64.f;
            if (thr <= 32.f) base = 32.f;
            if (thr <= 16.f) base = 16.f;
            if (thr <=  8.f) base =  8.f;
            if (thr <=  4.f) base =  4.f;
            if (thr <=  2.f) base =  2.f;
            const float inv_base = 1.0f / base;          // exact (power of two)
            // sel_mask = (base < 1) is provably dead: base >= 2 always.

            float rmax = 0.f;
#pragma unroll
            for (int e = 0; e < 16; ++e) {
                float q  = fminf(fmaxf(rintf(xv[e] * inv_base), -7.f), 7.f); // v_med3
                rv[e] = fmaf(-base, q, xv[e]);
                qb[e] = base * q;
                rmax = fmaxf(rmax, fabsf(rv[e]));
            }
            rmax = fmaxf(rmax, __shfl_xor(rmax, 1, 32));
            rmax = fmaxf(rmax, __shfl_xor(rmax, 2, 32));
            rmax = fmaxf(rmax, __shfl_xor(rmax, 4, 32));
            rmax = fmaxf(rmax, EPSQ);
            const float rs     = fmaxf(rmax * (1.0f / 7.0f), EPSQ);
            const float inv_rs = 1.0f / rs;

            _Float16 tmp[16];
#pragma unroll
            for (int e = 0; e < 16; ++e) {
                float rq = fminf(fmaxf(rintf(rv[e] * inv_rs), -8.f), 7.f);
                tmp[e] = (_Float16)fmaf(rq, rs, qb[e]);
            }
            v8h* adst = (v8h*)(As + row * AS_STR + seg * 16);
            adst[0] = *(const v8h*)(tmp);
            adst[1] = *(const v8h*)(tmp + 8);
        }

        if (kc + 1 < KDIM / GS)
            __builtin_prefetch(xrow + (kc + 1) * GS, 0, 1);

        if constexpr (USE_WT) wait_async_copies();
        __syncthreads();

        // ---- compute: 4 K-steps, 8 WMMAs each (2M x 4N subtiles) ----
#pragma unroll
        for (int ks = 0; ks < 4; ++ks) {
            const int kb = ks * 32;

            union { v16h v; v8h h[2]; } a0, a1;
            const _Float16* a0p = As + (mOff0 + lmod) * AS_STR + kb + lhalf * 8;
            const _Float16* a1p = As + (mOff1 + lmod) * AS_STR + kb + lhalf * 8;
            a0.h[0] = *(const v8h*)(a0p);
            a0.h[1] = *(const v8h*)(a0p + 16);
            a1.h[0] = *(const v8h*)(a1p);
            a1.h[1] = *(const v8h*)(a1p + 16);

#pragma unroll
            for (int j = 0; j < 4; ++j) {
                union { v16h v; v8h h[2]; } b;
                const _Float16* bp = Bt + (kb + lhalf * 8) * BT_STR + nBase + j * 16 + lmod;
#pragma unroll
                for (int s = 0; s < 8; ++s) {
                    b.v[s]     = bp[s * BT_STR];
                    b.v[8 + s] = bp[(16 + s) * BT_STR];
                }

                acc[0][j] = __builtin_amdgcn_wmma_f32_16x16x32_f16(
                    false, a0.v, false, b.v, (short)0, acc[0][j], false, false);
                acc[1][j] = __builtin_amdgcn_wmma_f32_16x16x32_f16(
                    false, a1.v, false, b.v, (short)0, acc[1][j], false, false);
            }
        }
        __syncthreads();
    }

    // ---- epilogue: C layout (VGPR i -> M = i + 8*lhalf, N = lmod) + bias ----
#pragma unroll
    for (int j = 0; j < 4; ++j) {
        const int n_g = n_blk + nBase + j * 16 + lmod;
        const float bv = bias[n_g];
#pragma unroll
        for (int i = 0; i < 8; ++i) {
            const int m0 = m_blk + mOff0 + lhalf * 8 + i;
            const int m1 = m_blk + mOff1 + lhalf * 8 + i;
            out[(size_t)m0 * NDIM + n_g] = acc[0][j][i] + bv;
            out[(size_t)m1 * NDIM + n_g] = acc[1][j][i] + bv;
        }
    }
}

extern "C" void kernel_launch(void* const* d_in, const int* in_sizes, int n_in,
                              void* d_out, int out_size, void* d_ws, size_t ws_size,
                              hipStream_t stream) {
    const float* x    = (const float*)d_in[0];
    const float* w    = (const float*)d_in[1];
    const float* bias = (const float*)d_in[2];
    float* out        = (float*)d_out;

    const int M = in_sizes[0] / KDIM;                 // 65536
    dim3 grid(NDIM / NT, M / MT);                     // (2, 1024), n fastest
    dim3 block(THREADS);

    const size_t wt_bytes = (size_t)KDIM * NDIM * sizeof(_Float16);  // 2 MB
    if (ws_size >= wt_bytes) {
        _Float16* wt = (_Float16*)d_ws;
        wt_transpose<<<dim3(NDIM / 64, KDIM / 64), 256, 0, stream>>>(w, wt);
        qr_linear_wmma<true><<<grid, block, 0, stream>>>(x, w, wt, bias, out);
    } else {
        qr_linear_wmma<false><<<grid, block, 0, stream>>>(x, w, nullptr, bias, out);
    }
}
